// GAT_51788715655650
// MI455X (gfx1250) — matrix operand order repaired and checked by reading, not compiled, not run on hardware
//
#include <hip/hip_runtime.h>
#include <math.h>

// ---------------------------------------------------------------------------
// GAT forward on gfx1250: WMMA f16 GEMMs + L2-resident atomic edge aggregation
// ---------------------------------------------------------------------------

typedef __attribute__((ext_vector_type(16))) _Float16 v16h;
typedef __attribute__((ext_vector_type(8)))  float    v8f;

#define NEG_SLOPE 0.2f

// ---------------------------------------------------------------------------
// GEMM: C[n, KOUT] = A[n,128] @ W[128,KOUT] (+bias). KOUT = NCT*16.
// One wave computes a 16 x KOUT strip via NCT accumulators, K in 4 chunks of 32.
// W is converted f32->f16 and PRE-PACKED into the WMMA B-fragment layout in
// LDS, so each fragment read is one aligned 32-byte LDS access (2x ds_load_b128)
// instead of 16 strided ds_load_u16.
//   bfrag[(kc*NCT + c)*32 + lane][j]  <->  W[(kc*32 + hi*16 + j), c*16 + n]
//   with lane = hi*16 + n
// ---------------------------------------------------------------------------
template <int NCT>
__global__ __launch_bounds__(256) void wmma_gemm(
    const float* __restrict__ A, const float* __restrict__ W,
    const float* __restrict__ bias, float* __restrict__ C, int nrows) {
  constexpr int KOUT = NCT * 16;
  __shared__ v16h bfrag[4 * NCT * 32];
  _Float16* wl = (_Float16*)bfrag;

  // cooperative f32 -> f16 conversion + fragment packing (coalesced reads)
  for (int idx = threadIdx.x; idx < 128 * KOUT; idx += 256) {
    const int k   = idx / KOUT;        // 0..127
    const int col = idx % KOUT;
    const int kc  = k >> 5;            // k-chunk
    const int kr  = k & 31;
    const int bhi = kr >> 4;
    const int j   = kr & 15;
    const int c   = col >> 4;
    const int bn  = col & 15;
    const int ln  = bhi * 16 + bn;
    wl[(((kc * NCT + c) * 32) + ln) * 16 + j] = (_Float16)W[idx];
  }
  __syncthreads();

  const int wave = threadIdx.x >> 5;   // 0..7
  const int lane = threadIdx.x & 31;
  const int n    = lane & 15;          // col within 16-tile / row-id for A
  const int hi   = lane >> 4;          // lane half select

  const long tile = (long)blockIdx.x * 8 + wave;
  const long row0 = tile * 16;
  int arow = (int)row0 + n;
  if (arow >= nrows) arow = nrows - 1;   // clamp loads; stores are guarded

  v8f acc[NCT] = {};

#pragma unroll
  for (int kc = 0; kc < 4; ++kc) {
    // --- A fragment (16x32 f16, wave32 layout) ---
    // elements 0..15 <-> K = kc*32 + hi*16 + j
    const float* ap = A + (long)arow * 128 + kc * 32 + hi * 16;
    const float4 p0 = *(const float4*)(ap + 0);
    const float4 p1 = *(const float4*)(ap + 4);
    const float4 p2 = *(const float4*)(ap + 8);
    const float4 p3 = *(const float4*)(ap + 12);
    v16h a;
    a[0]  = (_Float16)p0.x; a[1]  = (_Float16)p0.y;
    a[2]  = (_Float16)p0.z; a[3]  = (_Float16)p0.w;
    a[4]  = (_Float16)p1.x; a[5]  = (_Float16)p1.y;
    a[6]  = (_Float16)p1.z; a[7]  = (_Float16)p1.w;
    a[8]  = (_Float16)p2.x; a[9]  = (_Float16)p2.y;
    a[10] = (_Float16)p2.z; a[11] = (_Float16)p2.w;
    a[12] = (_Float16)p3.x; a[13] = (_Float16)p3.y;
    a[14] = (_Float16)p3.z; a[15] = (_Float16)p3.w;

#pragma unroll
    for (int c = 0; c < NCT; ++c) {
      const v16h b = bfrag[(kc * NCT + c) * 32 + lane];
      acc[c] = __builtin_amdgcn_wmma_f32_16x16x32_f16(
          false, a, false, b, (short)0, acc[c], false, false);
    }
  }

  // --- store D: lane half hi -> rows row0 + hi*8 + v, col = c*16 + n ---
#pragma unroll
  for (int v = 0; v < 8; ++v) {
    const int r = (int)row0 + hi * 8 + v;
    if (r < nrows) {
#pragma unroll
      for (int c = 0; c < NCT; ++c) {
        float val = acc[c][v];
        if (bias) val += bias[c * 16 + n];
        C[(long)r * KOUT + c * 16 + n] = val;
      }
    }
  }
}

// ---------------------------------------------------------------------------
// Per-(node, head) attention coefficients: es = <h, a_src>, ed = <h, a_dst>
// ---------------------------------------------------------------------------
__global__ void attn_coeff(const float* __restrict__ hp,
                           const float* __restrict__ a_src,
                           const float* __restrict__ a_dst,
                           float* __restrict__ es, float* __restrict__ ed,
                           int n, int heads, int ch) {
  int i = blockIdx.x * blockDim.x + threadIdx.x;
  if (i >= n * heads) return;
  const int node = i / heads, h = i % heads;
  const float* hv = hp + (long)node * heads * ch + h * ch;
  const float* av = a_src + h * ch;
  const float* dv = a_dst + h * ch;
  float s = 0.f, d = 0.f;
  for (int c = 0; c < ch; ++c) {
    const float x = hv[c];
    s += x * av[c];
    d += x * dv[c];
  }
  es[i] = s;
  ed[i] = d;
}

// ---------------------------------------------------------------------------
__global__ void fill_f32(float* __restrict__ p, long n, float v) {
  long i = (long)blockIdx.x * blockDim.x + threadIdx.x;
  if (i < n) p[i] = v;
}

__device__ __forceinline__ float leaky(float v) {
  return v > 0.f ? v : NEG_SLOPE * v;
}

// float atomic max via int/uint monotonic mapping
__device__ __forceinline__ void atomicMaxFloat(float* addr, float val) {
  if (val >= 0.f)
    atomicMax((int*)addr, __float_as_int(val));
  else
    atomicMin((unsigned int*)addr, __float_as_uint(val));
}

// ---------------------------------------------------------------------------
// Edge pass 1: segment max of leaky_relu(es[src] + ed[dst]) over dst
// ---------------------------------------------------------------------------
__global__ void edge_max(const int* __restrict__ ei,
                         const float* __restrict__ es,
                         const float* __restrict__ ed,
                         float* __restrict__ m, int etot, int heads) {
  int e = blockIdx.x * blockDim.x + threadIdx.x;
  if (e >= etot) return;
  const int s = ei[e], d = ei[etot + e];
  for (int h = 0; h < heads; ++h) {
    const float v = leaky(es[s * heads + h] + ed[d * heads + h]);
    atomicMaxFloat(&m[d * heads + h], v);
  }
}

// ---------------------------------------------------------------------------
// Edge pass 2: den[dst] += exp(e - m[dst])
// ---------------------------------------------------------------------------
__global__ void edge_den(const int* __restrict__ ei,
                         const float* __restrict__ es,
                         const float* __restrict__ ed,
                         const float* __restrict__ m,
                         float* __restrict__ den, int etot, int heads) {
  int e = blockIdx.x * blockDim.x + threadIdx.x;
  if (e >= etot) return;
  const int s = ei[e], d = ei[etot + e];
  for (int h = 0; h < heads; ++h) {
    const float v = leaky(es[s * heads + h] + ed[d * heads + h]);
    atomicAdd(&den[d * heads + h], __expf(v - m[d * heads + h]));
  }
}

// ---------------------------------------------------------------------------
// Edge pass 3: agg[dst, :] += hp[src, :] * alpha  (one wave32 per edge)
// heads*ch = 128 -> 4 channels/lane, or 64 -> 2 channels/lane
// ---------------------------------------------------------------------------
__global__ __launch_bounds__(256) void edge_agg(
    const int* __restrict__ ei, const float* __restrict__ es,
    const float* __restrict__ ed, const float* __restrict__ m,
    const float* __restrict__ den, const float* __restrict__ hp,
    float* __restrict__ agg, int etot, int heads, int ch) {
  const int wid  = (blockIdx.x * blockDim.x + threadIdx.x) >> 5;
  const int lane = threadIdx.x & 31;
  if (wid >= etot) return;
  const int s = ei[wid], d = ei[etot + wid];
  const int hh  = heads * ch;   // 128 or 64
  const int cpl = hh >> 5;      // 4 or 2 channels per lane
  const int c0  = lane * cpl;
  const int h   = c0 / ch;      // one head per lane (cpl <= ch always)

  const float v = leaky(es[s * heads + h] + ed[d * heads + h]);
  const float alpha = __expf(v - m[d * heads + h]) / den[d * heads + h];

  const float* hs = hp + (long)s * hh + c0;
  float* op = agg + (long)d * hh + c0;
  for (int c = 0; c < cpl; ++c) atomicAdd(&op[c], hs[c] * alpha);
}

// ---------------------------------------------------------------------------
// Epilogue: x = (relu?) (x + bias[col])
// ---------------------------------------------------------------------------
__global__ void bias_act(float* __restrict__ x, const float* __restrict__ b,
                         long n, int width, int do_relu) {
  long i = (long)blockIdx.x * blockDim.x + threadIdx.x;
  if (i >= n) return;
  float v = x[i] + b[(int)(i % width)];
  if (do_relu) v = fmaxf(v, 0.f);
  x[i] = v;
}

// ---------------------------------------------------------------------------
extern "C" void kernel_launch(void* const* d_in, const int* in_sizes, int n_in,
                              void* d_out, int out_size, void* d_ws,
                              size_t ws_size, hipStream_t stream) {
  const float* x     = (const float*)d_in[0];
  const float* enc_w = (const float*)d_in[2];
  const float* enc_b = (const float*)d_in[3];
  const float* W1  = (const float*)d_in[4];
  const float* as1 = (const float*)d_in[5];
  const float* ad1 = (const float*)d_in[6];
  const float* b1  = (const float*)d_in[7];
  const float* W2  = (const float*)d_in[8];
  const float* as2 = (const float*)d_in[9];
  const float* ad2 = (const float*)d_in[10];
  const float* b2  = (const float*)d_in[11];
  const float* W3  = (const float*)d_in[12];
  const float* as3 = (const float*)d_in[13];
  const float* ad3 = (const float*)d_in[14];
  const float* b3  = (const float*)d_in[15];
  const int*   ei  = (const int*)d_in[16];

  const int N    = in_sizes[0] / 128;   // 100000
  const int Etot = in_sizes[16] / 2;    // E + N
  float* out = (float*)d_out;

  // workspace layout
  float* h   = (float*)d_ws;            // [N,128] layer input / aggregation
  float* hp  = h  + (size_t)N * 128;    // [N,128] projected features
  float* es  = hp + (size_t)N * 128;    // [N,4]
  float* ed  = es + (size_t)N * 4;
  float* mb  = ed + (size_t)N * 4;      // segment max
  float* den = mb + (size_t)N * 4;      // segment sum

  const int TB = 256;
  const int gemm_blocks = (int)(((long)(N + 15) / 16 + 7) / 8);
  const int edge_blocks = (Etot + TB - 1) / TB;
  const int wave_blocks = (Etot + 7) / 8;   // one wave32 per edge
  auto gcount = [&](long n) { return (int)((n + TB - 1) / TB); };

  // ---------------- encoder: h = x @ enc_w + enc_b ----------------
  wmma_gemm<8><<<gemm_blocks, TB, 0, stream>>>(x, enc_w, enc_b, h, N);

  // ---------------- GAT layers 1 & 2 (4 heads x 32 ch) ----------------
  const float* Wl[2]  = {W1, W2};
  const float* asl[2] = {as1, as2};
  const float* adl[2] = {ad1, ad2};
  const float* bl[2]  = {b1, b2};
  for (int l = 0; l < 2; ++l) {
    wmma_gemm<8><<<gemm_blocks, TB, 0, stream>>>(h, Wl[l], nullptr, hp, N);
    attn_coeff<<<gcount((long)N * 4), TB, 0, stream>>>(hp, asl[l], adl[l],
                                                       es, ed, N, 4, 32);
    fill_f32<<<gcount((long)N * 4), TB, 0, stream>>>(mb, (long)N * 4,
                                                     -INFINITY);
    fill_f32<<<gcount((long)N * 4), TB, 0, stream>>>(den, (long)N * 4, 0.f);
    edge_max<<<edge_blocks, TB, 0, stream>>>(ei, es, ed, mb, Etot, 4);
    edge_den<<<edge_blocks, TB, 0, stream>>>(ei, es, ed, mb, den, Etot, 4);
    fill_f32<<<gcount((long)N * 128), TB, 0, stream>>>(h, (long)N * 128, 0.f);
    edge_agg<<<wave_blocks, TB, 0, stream>>>(ei, es, ed, mb, den, hp, h,
                                             Etot, 4, 32);
    bias_act<<<gcount((long)N * 128), TB, 0, stream>>>(h, bl[l],
                                                       (long)N * 128, 128, 1);
  }

  // ---------------- GAT layer 3 (1 head x 64 ch) -> d_out ----------------
  wmma_gemm<4><<<gemm_blocks, TB, 0, stream>>>(h, W3, nullptr, hp, N);
  attn_coeff<<<gcount((long)N), TB, 0, stream>>>(hp, as3, ad3, es, ed,
                                                 N, 1, 64);
  fill_f32<<<gcount((long)N), TB, 0, stream>>>(mb, (long)N, -INFINITY);
  fill_f32<<<gcount((long)N), TB, 0, stream>>>(den, (long)N, 0.f);
  edge_max<<<edge_blocks, TB, 0, stream>>>(ei, es, ed, mb, Etot, 1);
  edge_den<<<edge_blocks, TB, 0, stream>>>(ei, es, ed, mb, den, Etot, 1);
  fill_f32<<<gcount((long)N * 64), TB, 0, stream>>>(out, (long)N * 64, 0.f);
  edge_agg<<<wave_blocks, TB, 0, stream>>>(ei, es, ed, mb, den, hp, out,
                                           Etot, 1, 64);
  bias_act<<<gcount((long)N * 64), TB, 0, stream>>>(out, b3, (long)N * 64,
                                                    64, 0);
}
